// GraphPool_28157805593351
// MI455X (gfx1250) — matrix operand order repaired and checked by reading, not compiled, not run on hardware
//
#include <hip/hip_runtime.h>

typedef __attribute__((ext_vector_type(16))) _Float16       v16h;
typedef __attribute__((ext_vector_type(8)))  float          v8f;
typedef __attribute__((ext_vector_type(4)))  int            v4i;
typedef __attribute__((ext_vector_type(2)))  unsigned short u16x2;
typedef __attribute__((ext_vector_type(8)))  unsigned int   u32x8;

#define N_NODES 16384
#define DDIM    128
#define KTILES  (N_NODES / 32)   // 512 K-tiles of 32
#define NTILES  (DDIM / 16)      // 8  N-tiles of 16
#define MROWS   32               // rows per strip (2 WMMA M-tiles -> B reuse x2)
#define STRIPS  (N_NODES / MROWS)// 512 strips
#define KSPLIT  4                // waves per strip (K split 4 ways)

// adj in {0,1}: f16 bits = a*0x3C00.  Two ints -> one VGPR of two f16:
// v_lshl_or_b32 + v_pk_mul_lo_u16.
static __device__ __forceinline__ unsigned cvt01x2(int lo, int hi) {
    unsigned p = (unsigned)lo | ((unsigned)hi << 16);
    u16x2 v = __builtin_bit_cast(u16x2, p);
    v = v * (unsigned short)0x3C00;
    return __builtin_bit_cast(unsigned, v);
}

// Build a 16x32 f16 A fragment from four nontemporal b128 loads of one adj row.
static __device__ __forceinline__ v16h loadA(const int* __restrict__ ap) {
    v4i a0 = __builtin_nontemporal_load((const v4i*)(ap +  0)); // K: g*8+0..3
    v4i a1 = __builtin_nontemporal_load((const v4i*)(ap +  4)); // K: g*8+4..7
    v4i a2 = __builtin_nontemporal_load((const v4i*)(ap + 16)); // K: 16+g*8+0..3
    v4i a3 = __builtin_nontemporal_load((const v4i*)(ap + 20)); // K: 16+g*8+4..7
    u32x8 au;
    au[0] = cvt01x2(a0[0], a0[1]);  au[1] = cvt01x2(a0[2], a0[3]);
    au[2] = cvt01x2(a1[0], a1[1]);  au[3] = cvt01x2(a1[2], a1[3]);
    au[4] = cvt01x2(a2[0], a2[1]);  au[5] = cvt01x2(a2[2], a2[3]);
    au[6] = cvt01x2(a3[0], a3[1]);  au[7] = cvt01x2(a3[2], a3[3]);
    return __builtin_bit_cast(v16h, au);
}

// ---------------------------------------------------------------------------
// Pre-pass: x (fp32 row-major) -> f16 fragments in WMMA B-operand layout
// (32x16 KxN tile; lane group g = lane>>4 covers K = g*8+(e&7)+16*(e>>3), n = lane&15).
// ---------------------------------------------------------------------------
__global__ __launch_bounds__(256)
void pack_x_kernel(const float* __restrict__ x, _Float16* __restrict__ xh) {
    int tid  = blockIdx.x * blockDim.x + threadIdx.x;   // 16384*128 threads
    int f    = tid >> 9;
    int idx  = tid & 511;
    int lane = idx >> 4;
    int e    = idx & 15;
    int kt   = f >> 3;
    int u    = f & 7;
    int k    = kt * 32 + (lane >> 4) * 8 + (e & 7) + (e >> 3) * 16;
    int n    = u * 16 + (lane & 15);
    xh[tid]  = (_Float16)x[(size_t)k * DDIM + n];
}

// ---------------------------------------------------------------------------
// Main GEMM: out = x + adj@x.  One 32-row strip per block; 4 waves split K.
// Each wave: 2 A fragments x 8 B fragments -> 16 WMMAs / K-tile, 16 v8f
// accumulators.  Deterministic LDS reduction (64 KB) + coalesced epilogue.
// ---------------------------------------------------------------------------
__global__ __launch_bounds__(128)
void graph_pool_gemm(const int* __restrict__ adj, const _Float16* __restrict__ xh,
                     const float* __restrict__ x, float* __restrict__ out) {
    __shared__ float part[KSPLIT][MROWS * DDIM];        // 4 * 16 KB = 64 KB

    const int lane  = threadIdx.x & 31;
    const int split = threadIdx.x >> 5;                 // 0..3 (wave id)
    const int strip = blockIdx.x;                       // 0..511
    const int g     = lane >> 4;                        // lane group (ISA layout)
    const int m     = lane & 15;                        // A row / B,C column

    const int ktBeg = split * (KTILES / KSPLIT);
    const int ktEnd = ktBeg + (KTILES / KSPLIT);

    // rows this lane feeds: strip*32 + m (tile 0) and strip*32 + 16 + m (tile 1)
    const int* __restrict__ arow0 = adj + (size_t)(strip * MROWS + m) * N_NODES;
    const int* __restrict__ arow1 = arow0 + (size_t)16 * N_NODES;

    v8f c0[NTILES], c1[NTILES];
#pragma unroll
    for (int u = 0; u < NTILES; ++u) { c0[u] = (v8f)0.0f; c1[u] = (v8f)0.0f; }

    for (int kt = ktBeg; kt < ktEnd; ++kt) {
        const int koff = kt * 32 + g * 8;
        v16h A0 = loadA(arow0 + koff);
        v16h A1 = loadA(arow1 + koff);
        __builtin_prefetch(arow0 + koff + 64, 0, 0);    // global_prefetch_b8
        __builtin_prefetch(arow1 + koff + 64, 0, 0);

        const _Float16* __restrict__ bp =
            xh + (size_t)kt * NTILES * 512 + lane * 16;
#pragma unroll
        for (int u = 0; u < NTILES; ++u) {
            v16h B = *(const v16h*)(bp + (size_t)u * 512);   // L2-resident
            c0[u] = __builtin_amdgcn_wmma_f32_16x16x32_f16(
                false, A0, false, B, (short)0, c0[u], false, false);
            c1[u] = __builtin_amdgcn_wmma_f32_16x16x32_f16(
                false, A1, false, B, (short)0, c1[u], false, false);
        }
    }

    // Dump partial 32x128 tile.  C/D layout: VGPR r -> M = g*8 + r, N = m.
#pragma unroll
    for (int u = 0; u < NTILES; ++u) {
#pragma unroll
        for (int r = 0; r < 8; ++r) {
            part[split][(g * 8 + r) * DDIM + u * 16 + m]        = c0[u][r];
            part[split][(16 + g * 8 + r) * DDIM + u * 16 + m]   = c1[u][r];
        }
    }
    __syncthreads();

    // Deterministic reduce + "+x": 128 threads over 32*128 elements, coalesced.
#pragma unroll
    for (int i = 0; i < MROWS * DDIM / 128; ++i) {
        int e   = i * 128 + threadIdx.x;    // row_local*128 + col
        float s = part[0][e] + part[1][e] + part[2][e] + part[3][e];
        int row = strip * MROWS + (e >> 7);
        int col = e & (DDIM - 1);
        size_t o = (size_t)row * DDIM + col;
        out[o] = x[o] + s;
    }
}

extern "C" void kernel_launch(void* const* d_in, const int* in_sizes, int n_in,
                              void* d_out, int out_size, void* d_ws, size_t ws_size,
                              hipStream_t stream) {
    const float* x   = (const float*)d_in[0];      // [16384,128] fp32
    const int*   adj = (const int*)d_in[1];        // [16384,16384] int32 {0,1}
    float*       out = (float*)d_out;              // [16384,128] fp32
    _Float16*    xh  = (_Float16*)d_ws;            // 4 MiB packed B fragments

    pack_x_kernel<<<(N_NODES * DDIM) / 256, 256, 0, stream>>>(x, xh);

    // 512 strips (32 rows each), 4 K-split waves per block -> 2048 waves
    graph_pool_gemm<<<STRIPS, 128, 0, stream>>>(adj, xh, x, out);
}
